// FrequencyMaskingLoss_1692217115486
// MI455X (gfx1250) — matrix operand order repaired and checked by reading, not compiled, not run on hardware
//
#include <hip/hip_runtime.h>
#include <hip/hip_bf16.h>
#include <cstdint>

#define WIN      512
#define HOP      128
#define NF       257          // rfft bins
#define NCOL     514          // interleaved re/im columns
#define NFRAMES  1247
#define MT       78           // 78*16 = 1248 padded frames
#define NT       33           // 33*16 = 528 padded columns
#define NPAD     (NT*16)      // 528
#define LSIG     160000
#define LPAD     160512       // pad so last (invalid) padded frame reads in-bounds

typedef __attribute__((ext_vector_type(16))) _Float16 v16h;
typedef __attribute__((ext_vector_type(8)))  _Float16 v8h;
typedef __attribute__((ext_vector_type(8)))  float    v8f;

__device__ __forceinline__ unsigned enc_f(float x) {
    unsigned u = __float_as_uint(x);
    return (u >> 31) ? ~u : (u | 0x80000000u);
}
__device__ __forceinline__ float dec_f(unsigned u) {
    u = (u >> 31) ? (u & 0x7FFFFFFFu) : ~u;
    return __uint_as_float(u);
}

// ---------------- prep kernels ----------------

__global__ void k_init(unsigned* pmax) {
    if (threadIdx.x == 0 && blockIdx.x == 0) pmax[0] = 0u;
}

__global__ void k_prep_sig(const float* __restrict__ xr, const float* __restrict__ xa,
                           _Float16* __restrict__ xh, _Float16* __restrict__ dh) {
    int i = blockIdx.x * blockDim.x + threadIdx.x;
    if (i >= LPAD) return;
    float r = (i < LSIG) ? xr[i] : 0.0f;
    float a = (i < LSIG) ? xa[i] : 0.0f;
    xh[i] = (_Float16)r;
    dh[i] = (_Float16)(a - r);
}

// Bt[n][k] = hann[k] * (n even: cos(2*pi*f*k/512), n odd: -sin(...)), f = n/2; zero for n>=514
__global__ void k_prep_dft(_Float16* __restrict__ Bt) {
    int idx = blockIdx.x * blockDim.x + threadIdx.x;
    if (idx >= NPAD * WIN) return;
    int n = idx / WIN, k = idx % WIN;
    float val = 0.0f;
    if (n < NCOL) {
        int f = n >> 1;
        float hann = 0.5f * (1.0f - cosf((float)k * (6.28318530717958647692f / (float)WIN)));
        int m = (f * k) & (WIN - 1);                 // exact angle reduction mod 2*pi
        float th = (float)m * (6.28318530717958647692f / (float)WIN);
        float s, c;
        sincosf(th, &s, &c);
        val = hann * ((n & 1) ? -s : c);
    }
    Bt[idx] = (_Float16)val;
}

__global__ void k_prep_const(float* __restrict__ bark, float* __restrict__ ath_db,
                             float* __restrict__ ath_pow, float* __restrict__ dshift) {
    int f = blockIdx.x * blockDim.x + threadIdx.x;
    if (f >= NF) return;
    float fr = (float)f * (8000.0f / 256.0f);
    float q  = fr / 7500.0f;
    float b  = 13.0f * atanf(0.00076f * fr) + 3.5f * atanf(q * q);
    bark[f]   = b;
    dshift[f] = -6.025f - 0.275f * b;
    float ath;
    if (fr >= 20.0f && fr <= 20000.0f) {
        float fk = fr * 0.001f;
        ath = 3.64f * powf(fk, -0.8f)
            - 6.5f * expf(-0.6f * (fk - 3.3f) * (fk - 3.3f))
            + 0.001f * fk * fk * fk * fk - 12.0f;
    } else {
        ath = -INFINITY;                             // only bin 0 (0 Hz)
    }
    ath_db[f]  = ath;
    ath_pow[f] = (f == 0) ? 0.0f : exp10f(ath * 0.1f);
}

// ---------------- WMMA STFT (DFT-as-matmul) + fused PSD epilogue ----------------
// One wave per 16x16 output tile. K = 512 in 16 chunks of 32 (v_wmma_f32_16x16x32_f16).
// mode 0: out = PSD dB [1247*257], update encoded global max
// mode 1: out = linear (GAIN/WIN)^2 * |S|^2 [1247*257]

__global__ __launch_bounds__(32)
void k_stft_wmma(const _Float16* __restrict__ sig, const _Float16* __restrict__ Bt,
                 float* __restrict__ out, unsigned* __restrict__ pmax, int mode) {
    const int lane = threadIdx.x & 31;
    const int hl   = lane >> 4;          // half-wave select
    const int mn   = lane & 15;          // A row / B column within tile
    const int tn   = blockIdx.x;         // column tile
    const int tm   = blockIdx.y;         // frame tile

    const _Float16* ap = sig + (size_t)(tm * 16 + mn) * HOP;   // frame start (window folded into B)
    const _Float16* bp = Bt  + (size_t)(tn * 16 + mn) * WIN;   // DFT matrix row (n-major)

    v8f acc = {0.f, 0.f, 0.f, 0.f, 0.f, 0.f, 0.f, 0.f};
    #pragma unroll
    for (int kc = 0; kc < WIN / 32; ++kc) {
        const int k0 = kc * 32;
        // A (16x32 f16): lane element e -> K = hl*16+e (contiguous, matches B)
        v8h alo = *(const v8h*)(ap + k0 + hl * 16);
        v8h ahi = *(const v8h*)(ap + k0 + hl * 16 + 8);
        v16h a;
        #pragma unroll
        for (int i = 0; i < 8; ++i) { a[i] = alo[i]; a[i + 8] = ahi[i]; }
        // B (32x16 f16): lanes 0-15 hold K=0..15, lanes 16-31 hold K=16..31 (contiguous 32B)
        v16h b = *(const v16h*)(bp + k0 + hl * 16);
        acc = __builtin_amdgcn_wmma_f32_16x16x32_f16(false, a, false, b,
                                                     (short)0, acc, false, false);
    }

    // C layout: VGPR v -> row = hl*8+v; N = mn. Adjacent columns are (re, im) of one bin.
    const int   n_glob = tn * 16 + mn;
    const float k2     = (8.0f / 3.0f) / ((float)WIN * (float)WIN);   // GAIN^2/WIN^2
    float wmax = -1e30f;
    #pragma unroll
    for (int v = 0; v < 8; ++v) {
        float re = acc[v];
        float im = __shfl_xor(re, 1);                 // partner column (all lanes execute)
        int row  = tm * 16 + hl * 8 + v;
        if (((mn & 1) == 0) && (n_glob < NCOL) && (row < NFRAMES)) {
            int   f    = n_glob >> 1;
            float mag2 = re * re + im * im;
            if (mode == 0) {
                float psd = 10.0f * log10f(mag2 * k2);    // log10f(0) = -inf -> clamp
                psd = fmaxf(psd, -200.0f);
                out[row * NF + f] = psd;
                wmax = fmaxf(wmax, psd);
            } else {
                out[row * NF + f] = mag2 * k2;
            }
        }
    }
    if (mode == 0) {
        #pragma unroll
        for (int off = 16; off >= 1; off >>= 1)
            wmax = fmaxf(wmax, __shfl_xor(wmax, off));
        if (lane == 0) atomicMax(pmax, enc_f(wmax));
    }
}

// ---------------- per-frame masking threshold + loss partials ----------------

__global__ __launch_bounds__(256)
void k_threshold_loss(const float* __restrict__ psd, const float* __restrict__ psd_d,
                      const float* __restrict__ bark, const float* __restrict__ ath_db,
                      const float* __restrict__ ath_pow, const float* __restrict__ dshift,
                      const unsigned* __restrict__ pmax, float* __restrict__ frame_loss) {
    const int t   = blockIdx.x;
    const int tid = threadIdx.x;
    __shared__ float sp[NF], spw[NF], sm[NF], mlist[NF];
    __shared__ int   ord[NF];
    __shared__ unsigned char vflag[NF], keep[NF];
    __shared__ int   s_n;
    __shared__ float red[256];

    const float psdmax = dec_f(*pmax);
    const float shift  = 96.0f - psdmax;

    for (int f = tid; f < NF; f += 256) {
        float p = psd[t * NF + f] + shift;
        sp[f]  = p;
        spw[f] = exp10f(p * 0.1f);
    }
    __syncthreads();
    for (int f = tid; f < NF; f += 256) {
        bool ismax = (f >= 1) && (f <= NF - 2) && (sp[f] > sp[f - 1]) && (sp[f] > sp[f + 1]);
        int fm1 = (f == 0) ? NF - 1 : f - 1;          // roll() wrap (irrelevant at edges, faithful)
        int fp1 = (f == NF - 1) ? 0 : f + 1;
        float mall = 10.0f * log10f(spw[fm1] + spw[f] + spw[fp1]);
        sm[f]    = mall;
        vflag[f] = (ismax && (mall > ath_db[f])) ? 1 : 0;
    }
    __syncthreads();
    if (tid == 0) {
        int n = 0;                                    // stable compaction == argsort(~valid)
        for (int f = 0; f < NF; ++f)
            if (vflag[f]) { ord[n] = f; mlist[n] = sm[f]; ++n; }
        for (int i = n; i < NF; ++i) { mlist[i] = -200.0f; ord[i] = 0; }
        for (int i = 0; i < NF; ++i) keep[i] = 1;
        int ip = 0;                                   // sequential bark-dedup scan
        for (int i = 1; i < NF; ++i) {
            bool active       = (i < n);
            bool close        = (bark[i] - bark[ip]) < 0.5f;   // list-position indexing (faithful)
            bool prev_smaller = mlist[ip] < mlist[i];
            if (active && close) keep[prev_smaller ? ip : i] = 0;
            if (active) ip = close ? (prev_smaller ? ip + 1 : ip) : i;
        }
        s_n = n;
    }
    __syncthreads();

    const int   n     = s_n;
    const float scale = exp10f(9.6f - psdmax * 0.1f);  // 10^9.6 / 10^(psd_max/10)
    float loc = 0.0f;
    for (int f = tid; f < NF; f += 256) {
        float bz = bark[f];
        float sumpow = 0.0f;
        for (int k = 0; k < n; ++k) {
            if (!keep[k]) continue;
            float mk = mlist[k];
            int   j  = ord[k];
            float dz = bz - bark[j];
            float slope = (dz > 0.0f) ? (-27.0f + 0.37f * fmaxf(mk - 40.0f, 0.0f)) : 27.0f;
            float tt = mk + dshift[j] + slope * dz;
            sumpow += exp10f(tt * 0.1f);
        }
        float thrpow = sumpow + ath_pow[f];            // == 10^(thr/10), linear domain
        loc += fmaxf(scale * psd_d[t * NF + f] - thrpow, 0.0f);
    }
    red[tid] = loc;
    __syncthreads();
    for (int s = 128; s > 0; s >>= 1) {
        if (tid < s) red[tid] += red[tid + s];
        __syncthreads();
    }
    if (tid == 0) frame_loss[t] = red[0];
}

__global__ __launch_bounds__(256)
void k_final(const float* __restrict__ frame_loss, float* __restrict__ out) {
    __shared__ float red[256];
    float s = 0.0f;
    for (int t = threadIdx.x; t < NFRAMES; t += 256) s += frame_loss[t];
    red[threadIdx.x] = s;
    __syncthreads();
    for (int st = 128; st > 0; st >>= 1) {
        if (threadIdx.x < st) red[threadIdx.x] += red[threadIdx.x + st];
        __syncthreads();
    }
    if (threadIdx.x == 0)
        out[0] = 1e-6f * (red[0] / (float)(NFRAMES * NF));   // ALPHA * mean
}

// ---------------- host launch ----------------

static inline char* align_up(char* p, size_t a) {
    return (char*)(((uintptr_t)p + (a - 1)) & ~(uintptr_t)(a - 1));
}

extern "C" void kernel_launch(void* const* d_in, const int* in_sizes, int n_in,
                              void* d_out, int out_size, void* d_ws, size_t ws_size,
                              hipStream_t stream) {
    const float* x_adv = (const float*)d_in[0];
    const float* x_ref = (const float*)d_in[1];
    float* out = (float*)d_out;

    char* w = (char*)d_ws;
    auto carve = [&](size_t bytes) -> void* {
        w = align_up(w, 512);
        void* r = (void*)w;
        w += bytes;
        return r;
    };
    _Float16* xh      = (_Float16*)carve((size_t)LPAD * 2);
    _Float16* dh      = (_Float16*)carve((size_t)LPAD * 2);
    _Float16* Bt      = (_Float16*)carve((size_t)NPAD * WIN * 2);
    float*    psd     = (float*)carve((size_t)NFRAMES * NF * 4);
    float*    psd_d   = (float*)carve((size_t)NFRAMES * NF * 4);
    float*    bark    = (float*)carve(NF * 4);
    float*    ath_db  = (float*)carve(NF * 4);
    float*    ath_pow = (float*)carve(NF * 4);
    float*    dshift  = (float*)carve(NF * 4);
    float*    floss   = (float*)carve(NFRAMES * 4);
    unsigned* pmax    = (unsigned*)carve(4);
    (void)in_sizes; (void)n_in; (void)out_size; (void)ws_size;

    k_init<<<1, 32, 0, stream>>>(pmax);
    k_prep_sig<<<(LPAD + 255) / 256, 256, 0, stream>>>(x_ref, x_adv, xh, dh);
    k_prep_dft<<<(NPAD * WIN + 255) / 256, 256, 0, stream>>>(Bt);
    k_prep_const<<<(NF + 255) / 256, 256, 0, stream>>>(bark, ath_db, ath_pow, dshift);

    dim3 grid(NT, MT);
    k_stft_wmma<<<grid, 32, 0, stream>>>(xh, Bt, psd,   pmax, 0);   // reference spectrum -> PSD dB + max
    k_stft_wmma<<<grid, 32, 0, stream>>>(dh, Bt, psd_d, pmax, 1);   // delta spectrum -> linear power

    k_threshold_loss<<<NFRAMES, 256, 0, stream>>>(psd, psd_d, bark, ath_db, ath_pow,
                                                  dshift, pmax, floss);
    k_final<<<1, 256, 0, stream>>>(floss, out);
}
